// self_attention_83975200571952
// MI455X (gfx1250) — matrix operand order repaired and checked by reading, not compiled, not run on hardware
//
#include <hip/hip_runtime.h>
#include <hip/hip_bf16.h>

// ---------------------------------------------------------------------------
// Multi-head causal self-attention forward for MI455X (gfx1250, wave32, WMMA)
//   B=4, S=2048, D_IN=D_OUT=1024, H=16, HD=64
// qkv_proj (WMMA bf16, double-buffered LDS) -> flash attention (WMMA bf16,
// online softmax, async K staging) -> out_proj (WMMA bf16 + bias, async A
// staging, fp32 out).
// ---------------------------------------------------------------------------

typedef __bf16 bf16;
typedef __attribute__((ext_vector_type(16))) __bf16 bf16x16;
typedef __attribute__((ext_vector_type(8)))  float  f32x8;

constexpr int Bb   = 4;
constexpr int Ss   = 2048;
constexpr int Din  = 1024;
constexpr int Dout = 1024;
constexpr int Hh   = 16;
constexpr int Hd   = 64;
constexpr int Mrows = Bb * Ss;           // 8192

union AFrag { bf16x16 v; uint4 q[2]; };  // 16 bf16 = 32 B = 8 VGPRs
union CFrag { f32x8  v; float f[8];  };

__device__ __forceinline__ f32x8 wmma_bf16(bf16x16 a, bf16x16 b, f32x8 c) {
  return __builtin_amdgcn_wmma_f32_16x16x32_bf16(
      false, a, false, b, (short)0, c, false, false);
}

// ---- gfx1250 async global->LDS copy (ISA ch.10 async memory, ASYNCcnt) ----
// GVS addressing: 64-bit uniform base (SGPR pair) + per-lane 32-bit offset.
// VDST VGPR carries the per-lane LDS byte address.
__device__ __forceinline__ void async_ld_b128(unsigned lds_addr, unsigned voff,
                                              unsigned long long sbase) {
  asm volatile("global_load_async_to_lds_b128 %0, %1, %2 offset:0"
               :: "v"(lds_addr), "v"(voff), "s"(sbase) : "memory");
}
__device__ __forceinline__ void wait_async0() {
  asm volatile("s_wait_asynccnt 0x0" ::: "memory");
}
__device__ __forceinline__ unsigned lds_off(const void* p) {
  // Flat LDS addresses carry the LDS byte offset in the low 32 bits.
  return (unsigned)(uintptr_t)p;
}

// ---------------------------------------------------------------------------
// Kernel 1: QKV projection.  C[8192x1024] = X[8192x1024] @ W[1024x1024]
// Block = 256x128 tile, 8 waves as 4(M)x2(N); wave tile 64x64 = 4x4 WMMA
// fragments (16 wmma per 16 ds_load_b128). Double-buffered LDS staging.
// Output bf16 in [b][h][s][hd] layout.
// ---------------------------------------------------------------------------
__global__ void __launch_bounds__(256)
qkv_proj_kernel(const float* __restrict__ X,
                const float* __restrict__ Wq,
                const float* __restrict__ Wk,
                const float* __restrict__ Wv,
                bf16* __restrict__ Qo,
                bf16* __restrict__ Ko,
                bf16* __restrict__ Vo)
{
  __shared__ __align__(16) bf16 As[2][256 * 40];  // rows x 32K (stride 40)
  __shared__ __align__(16) bf16 Bs[2][128 * 40];  // cols x 32K (transposed)

  const int tid  = threadIdx.x;
  const int lane = tid & 31;
  const int wv   = tid >> 5;
  const int wm   = wv & 3;               // 64-row group
  const int wn   = wv >> 2;              // 64-col group
  const int m0   = blockIdx.y * 256;
  const int n0   = blockIdx.x * 128;

  const float* W  = (blockIdx.z == 0) ? Wq : (blockIdx.z == 1 ? Wk : Wv);
  bf16*       Dst = (blockIdx.z == 0) ? Qo : (blockIdx.z == 1 ? Ko : Vo);

  CFrag acc[4][4];
#pragma unroll
  for (int i = 0; i < 4; ++i)
#pragma unroll
    for (int j = 0; j < 4; ++j)
#pragma unroll
      for (int r = 0; r < 8; ++r) acc[i][j].f[r] = 0.0f;

  auto stageA = [&](bf16* dst, int k0) {
#pragma unroll
    for (int t = 0; t < 8; ++t) {
      int idx = tid + t * 256;            // 0..2047 float4 chunks
      int row = idx >> 3;                 // 0..255
      int c4  = idx & 7;
      float4 vx = *(const float4*)(X + (size_t)(m0 + row) * Din + k0 + c4 * 4);
      union { bf16 h[4]; uint2 u; } pk;
      pk.h[0] = (bf16)vx.x; pk.h[1] = (bf16)vx.y;
      pk.h[2] = (bf16)vx.z; pk.h[3] = (bf16)vx.w;
      *(uint2*)&dst[row * 40 + c4 * 4] = pk.u;
    }
  };
  auto stageB = [&](bf16* dst, int k0) {
#pragma unroll
    for (int t = 0; t < 4; ++t) {
      int idx = tid + t * 256;
      int kr  = idx >> 5;                 // 0..31
      int c4  = idx & 31;                 // float4 within 128 cols
      float4 vw = *(const float4*)(W + (size_t)(k0 + kr) * Dout + n0 + c4 * 4);
      dst[(c4 * 4 + 0) * 40 + kr] = (bf16)vw.x;
      dst[(c4 * 4 + 1) * 40 + kr] = (bf16)vw.y;
      dst[(c4 * 4 + 2) * 40 + kr] = (bf16)vw.z;
      dst[(c4 * 4 + 3) * 40 + kr] = (bf16)vw.w;
    }
  };

  stageA(As[0], 0);
  stageB(Bs[0], 0);
  __syncthreads();

  const int ln = lane & 15;
  const int kb = (lane < 16) ? 0 : 8;     // A-frag K base (ISA 7.12.2)
  const int kh = (lane < 16) ? 0 : 8;     // B-frag K base

  for (int kt = 0; kt < Din / 32; ++kt) {
    const int cur = kt & 1;
    if (kt + 1 < Din / 32) {              // stage next chunk while computing
      stageA(As[cur ^ 1], (kt + 1) * 32);
      stageB(Bs[cur ^ 1], (kt + 1) * 32);
    }
    AFrag a[4], b[4];
#pragma unroll
    for (int i = 0; i < 4; ++i) {
      const bf16* p = &As[cur][(wm * 64 + i * 16 + ln) * 40 + kb];
      a[i].q[0] = *(const uint4*)(p);
      a[i].q[1] = *(const uint4*)(p + 16);
    }
#pragma unroll
    for (int j = 0; j < 4; ++j) {
      const bf16* p = &Bs[cur][(wn * 64 + j * 16 + ln) * 40 + kh];
      b[j].q[0] = *(const uint4*)(p);
      b[j].q[1] = *(const uint4*)(p + 16);
    }
#pragma unroll
    for (int i = 0; i < 4; ++i)
#pragma unroll
      for (int j = 0; j < 4; ++j)
        acc[i][j].v = wmma_bf16(a[i].v, b[j].v, acc[i][j].v);
    __syncthreads();
  }

  const int rb = (lane < 16) ? 0 : 8;
#pragma unroll
  for (int i = 0; i < 4; ++i)
#pragma unroll
    for (int j = 0; j < 4; ++j) {
      int n  = n0 + wn * 64 + j * 16 + ln;
      int h  = n >> 6;
      int hd = n & 63;
#pragma unroll
      for (int r = 0; r < 8; ++r) {
        int m    = m0 + wm * 64 + i * 16 + r + rb;
        int bb   = m >> 11;
        int srow = m & 2047;
        Dst[((size_t)(bb * Hh + h) * Ss + srow) * Hd + hd] = (bf16)acc[i][j].f[r];
      }
    }
}

// ---------------------------------------------------------------------------
// Kernel 2: causal flash attention. Block = (b,h,64 query rows), 4 waves.
// K block staged with async global->LDS DMA; V staged transposed by VALU;
// prefetch of the next K/V block into L2.
// ---------------------------------------------------------------------------
__global__ void __launch_bounds__(128)
attn_kernel(const bf16* __restrict__ Qm, const bf16* __restrict__ Km,
            const bf16* __restrict__ Vm, bf16* __restrict__ Ctx)
{
  __shared__ __align__(16) bf16 Ks[64 * 72];       // K block [key][hd]
  __shared__ __align__(16) bf16 Vt[64 * 72];       // V block transposed [hd][key]
  __shared__ __align__(16) bf16 Pw[4][16 * 72];    // per-wave P tile [qrow][key]

  const int tid  = threadIdx.x;
  const int lane = tid & 31;
  const int w    = tid >> 5;
  const int qblk = blockIdx.x;
  const int h    = blockIdx.y;
  const int b    = blockIdx.z;
  const size_t bh = (size_t)(b * Hh + h) * Ss * Hd;
  const int q0   = qblk * 64 + w * 16;
  const int ln   = lane & 15;
  const int half = (lane < 16) ? 0 : 1;

  AFrag qa[2];
  {
    const int kb = half ? 8 : 0;
    const bf16* qrow = Qm + bh + (size_t)(q0 + ln) * Hd;
#pragma unroll
    for (int c = 0; c < 2; ++c) {
      qa[c].q[0] = *(const uint4*)(qrow + c * 32 + kb);
      qa[c].q[1] = *(const uint4*)(qrow + c * 32 + kb + 16);
    }
  }

  CFrag o[4];
  float mrow[8], lrow[8];
#pragma unroll
  for (int nt = 0; nt < 4; ++nt)
#pragma unroll
    for (int r = 0; r < 8; ++r) o[nt].f[r] = 0.0f;
#pragma unroll
  for (int r = 0; r < 8; ++r) { mrow[r] = -1e30f; lrow[r] = 0.0f; }

  const unsigned ks0 = lds_off(&Ks[0]);

  for (int kbi = 0; kbi <= qblk; ++kbi) {
    // --- stage K via async DMA, V via transpose scatter -------------------
    const bf16* kblk = Km + bh + (size_t)(kbi * 64) * Hd;
    const unsigned long long kb64 = (unsigned long long)(uintptr_t)kblk;
#pragma unroll
    for (int t = 0; t < 4; ++t) {
      int idx = tid + t * 128;            // 0..511 16B chunks
      int row = idx >> 3;                 // key row 0..63
      int c8  = idx & 7;                  // 16B chunk within 128B row
      async_ld_b128(ks0 + row * 144 + c8 * 16,
                    (unsigned)(row * 128 + c8 * 16), kb64);
      const size_t src = bh + (size_t)(kbi * 64 + row) * Hd + c8 * 8;
      union { uint4 u; bf16 hv[8]; } vv;
      vv.u = *(const uint4*)(Vm + src);
#pragma unroll
      for (int e = 0; e < 8; ++e) Vt[(c8 * 8 + e) * 72 + row] = vv.hv[e];
      if (kbi + 1 <= qblk) {              // prefetch next block into L2
        __builtin_prefetch(Km + src + (size_t)64 * Hd, 0, 1);
        __builtin_prefetch(Vm + src + (size_t)64 * Hd, 0, 1);
      }
    }
    wait_async0();
    __syncthreads();

    // --- scores: S = Q . K^T ---------------------------------------------
    CFrag s[4];
#pragma unroll
    for (int nt = 0; nt < 4; ++nt)
#pragma unroll
      for (int r = 0; r < 8; ++r) s[nt].f[r] = 0.0f;

    const int kh = half ? 8 : 0;
#pragma unroll
    for (int c = 0; c < 2; ++c)
#pragma unroll
      for (int nt = 0; nt < 4; ++nt) {
        AFrag bk;
        const bf16* p = &Ks[(nt * 16 + ln) * 72 + c * 32 + kh];
        bk.q[0] = *(const uint4*)(p);
        bk.q[1] = *(const uint4*)(p + 16);
        s[nt].v = wmma_bf16(qa[c].v, bk.v, s[nt].v);
      }

    const float scale = 0.125f;           // 1/sqrt(64)
    if (kbi == qblk) {
#pragma unroll
      for (int nt = 0; nt < 4; ++nt) {
        int key = kbi * 64 + nt * 16 + ln;
#pragma unroll
        for (int r = 0; r < 8; ++r) {
          int qg = q0 + r + half * 8;
          s[nt].f[r] = (key <= qg) ? s[nt].f[r] * scale : -1e30f;
        }
      }
    } else {
#pragma unroll
      for (int nt = 0; nt < 4; ++nt)
#pragma unroll
        for (int r = 0; r < 8; ++r) s[nt].f[r] *= scale;
    }

    // --- online softmax row stats (rows live in 16-lane halves) -----------
    float alpha[8];
#pragma unroll
    for (int r = 0; r < 8; ++r) {
      float mx = fmaxf(fmaxf(s[0].f[r], s[1].f[r]),
                       fmaxf(s[2].f[r], s[3].f[r]));
#pragma unroll
      for (int off = 1; off < 16; off <<= 1)
        mx = fmaxf(mx, __shfl_xor(mx, off, 32));
      float mnew = fmaxf(mrow[r], mx);
      alpha[r] = __expf(mrow[r] - mnew);
      mrow[r] = mnew;
      float rs = 0.0f;
#pragma unroll
      for (int nt = 0; nt < 4; ++nt) {
        float p = __expf(s[nt].f[r] - mnew);
        s[nt].f[r] = p;
        rs += p;
      }
#pragma unroll
      for (int off = 1; off < 16; off <<= 1)
        rs += __shfl_xor(rs, off, 32);
      lrow[r] = lrow[r] * alpha[r] + rs;
    }
#pragma unroll
    for (int nt = 0; nt < 4; ++nt)
#pragma unroll
      for (int r = 0; r < 8; ++r) o[nt].f[r] *= alpha[r];

    // --- P: C-layout -> per-wave LDS -> A-layout; O += P.V ----------------
    bf16* Pl = Pw[w];
#pragma unroll
    for (int nt = 0; nt < 4; ++nt)
#pragma unroll
      for (int r = 0; r < 8; ++r)
        Pl[(r + half * 8) * 72 + nt * 16 + ln] = (bf16)s[nt].f[r];

    const int kb2 = half ? 8 : 0;
#pragma unroll
    for (int c = 0; c < 2; ++c) {
      AFrag pa;
      const bf16* pp = &Pl[ln * 72 + c * 32 + kb2];
      pa.q[0] = *(const uint4*)(pp);
      pa.q[1] = *(const uint4*)(pp + 16);
#pragma unroll
      for (int nt = 0; nt < 4; ++nt) {
        AFrag vb;
        const bf16* vp = &Vt[(nt * 16 + ln) * 72 + c * 32 + kh];
        vb.q[0] = *(const uint4*)(vp);
        vb.q[1] = *(const uint4*)(vp + 16);
        o[nt].v = wmma_bf16(pa.v, vb.v, o[nt].v);
      }
    }
    __syncthreads();
  }

#pragma unroll
  for (int nt = 0; nt < 4; ++nt) {
    int hd = nt * 16 + ln;
#pragma unroll
    for (int r = 0; r < 8; ++r) {
      int srow = q0 + r + half * 8;
      float val = o[nt].f[r] / lrow[r];
      Ctx[((size_t)b * Ss + srow) * Dout + h * Hd + hd] = (bf16)val;
    }
  }
}

// ---------------------------------------------------------------------------
// Kernel 3: output projection.  out = ctx @ Wo + bo (fp32 out).
// 256x128 block tile, 64x64 wave tile, double-buffered; A tile (already
// bf16) staged via async global->LDS DMA.
// ---------------------------------------------------------------------------
__global__ void __launch_bounds__(256)
out_proj_kernel(const bf16* __restrict__ Ctx, const float* __restrict__ Wo,
                const float* __restrict__ bo, float* __restrict__ Out)
{
  __shared__ __align__(16) bf16 As[2][256 * 40];
  __shared__ __align__(16) bf16 Bs[2][128 * 40];

  const int tid  = threadIdx.x;
  const int lane = tid & 31;
  const int wv   = tid >> 5;
  const int wm   = wv & 3;
  const int wn   = wv >> 2;
  const int m0   = blockIdx.y * 256;
  const int n0   = blockIdx.x * 128;

  CFrag acc[4][4];
#pragma unroll
  for (int i = 0; i < 4; ++i)
#pragma unroll
    for (int j = 0; j < 4; ++j)
#pragma unroll
      for (int r = 0; r < 8; ++r) acc[i][j].f[r] = 0.0f;

  auto stageA = [&](bf16* dst, int k0) {   // async DMA: bf16 -> bf16
    const unsigned long long base =
        (unsigned long long)(uintptr_t)(Ctx + (size_t)m0 * Dout + k0);
    const unsigned d0 = lds_off(dst);
#pragma unroll
    for (int t = 0; t < 4; ++t) {
      int idx = tid + t * 256;             // 0..1023 16B chunks
      int row = idx >> 2;                  // 0..255
      int c   = idx & 3;                   // 16B chunk within 64B row
      async_ld_b128(d0 + row * 80 + c * 16,
                    (unsigned)(row * (Dout * 2) + c * 16), base);
    }
  };
  auto stageB = [&](bf16* dst, int k0) {
#pragma unroll
    for (int t = 0; t < 4; ++t) {
      int idx = tid + t * 256;
      int kr  = idx >> 5;
      int c4  = idx & 31;
      float4 vw = *(const float4*)(Wo + (size_t)(k0 + kr) * Dout + n0 + c4 * 4);
      dst[(c4 * 4 + 0) * 40 + kr] = (bf16)vw.x;
      dst[(c4 * 4 + 1) * 40 + kr] = (bf16)vw.y;
      dst[(c4 * 4 + 2) * 40 + kr] = (bf16)vw.z;
      dst[(c4 * 4 + 3) * 40 + kr] = (bf16)vw.w;
    }
  };

  stageA(As[0], 0);
  stageB(Bs[0], 0);
  wait_async0();
  __syncthreads();

  const int ln = lane & 15;
  const int kb = (lane < 16) ? 0 : 8;
  const int kh = (lane < 16) ? 0 : 8;

  for (int kt = 0; kt < Dout / 32; ++kt) {
    const int cur = kt & 1;
    if (kt + 1 < Dout / 32) {
      stageA(As[cur ^ 1], (kt + 1) * 32);
      stageB(Bs[cur ^ 1], (kt + 1) * 32);
    }
    AFrag a[4], b[4];
#pragma unroll
    for (int i = 0; i < 4; ++i) {
      const bf16* p = &As[cur][(wm * 64 + i * 16 + ln) * 40 + kb];
      a[i].q[0] = *(const uint4*)(p);
      a[i].q[1] = *(const uint4*)(p + 16);
    }
#pragma unroll
    for (int j = 0; j < 4; ++j) {
      const bf16* p = &Bs[cur][(wn * 64 + j * 16 + ln) * 40 + kh];
      b[j].q[0] = *(const uint4*)(p);
      b[j].q[1] = *(const uint4*)(p + 16);
    }
#pragma unroll
    for (int i = 0; i < 4; ++i)
#pragma unroll
      for (int j = 0; j < 4; ++j)
        acc[i][j].v = wmma_bf16(a[i].v, b[j].v, acc[i][j].v);
    wait_async0();
    __syncthreads();
  }

  const int rb = (lane < 16) ? 0 : 8;
#pragma unroll
  for (int i = 0; i < 4; ++i)
#pragma unroll
    for (int j = 0; j < 4; ++j) {
      int n = n0 + wn * 64 + j * 16 + ln;
      float bias = bo[n];
#pragma unroll
      for (int r = 0; r < 8; ++r) {
        int m = m0 + wm * 64 + i * 16 + r + rb;
        Out[(size_t)m * Dout + n] = acc[i][j].f[r] + bias;
      }
    }
}

// ---------------------------------------------------------------------------
extern "C" void kernel_launch(void* const* d_in, const int* in_sizes, int n_in,
                              void* d_out, int out_size, void* d_ws, size_t ws_size,
                              hipStream_t stream) {
  const float* x  = (const float*)d_in[0];
  const float* Wq = (const float*)d_in[1];
  const float* Wk = (const float*)d_in[2];
  const float* Wv = (const float*)d_in[3];
  const float* Wo = (const float*)d_in[4];
  const float* bo = (const float*)d_in[5];
  float* out = (float*)d_out;

  const size_t qsz = (size_t)Mrows * Dout;   // elements per bf16 buffer
  bf16* Q = (bf16*)d_ws;
  bf16* K = Q + qsz;
  bf16* V = K + qsz;
  bf16* C = V + qsz;                          // 64 MB workspace total

  dim3 gA(Dout / 128, Mrows / 256, 3);        // 8 x 32 x 3
  qkv_proj_kernel<<<gA, dim3(256), 0, stream>>>(x, Wq, Wk, Wv, Q, K, V);

  dim3 gB(Ss / 64, Hh, Bb);                   // 32 x 16 x 4
  attn_kernel<<<gB, dim3(128), 0, stream>>>(Q, K, V, C);

  dim3 gC(Dout / 128, Mrows / 256);           // 8 x 32
  out_proj_kernel<<<gC, dim3(256), 0, stream>>>(C, Wo, bo, out);
}
